// NetBurgers1D_InN_legendre_80865644249644
// MI455X (gfx1250) — matrix operand order repaired and checked by reading, not compiled, not run on hardware
//
#include <hip/hip_runtime.h>
#include <hip/hip_bf16.h>
#include <math.h>

// ---------------------------------------------------------------------------
// Problem constants (from the reference)
// ---------------------------------------------------------------------------
constexpr int B_N     = 32;
constexpr int IN_LEN  = 10;
constexpr int L_IN    = 16384;
constexpr int N_POLY  = 12;     // Legendre kernel length
constexpr int M_MODES = 6;
constexpr int CH      = 20;     // channels
constexpr int CHP     = 32;     // channels padded to WMMA K tile
constexpr int NB      = 4;      // blocks
constexpr int S_STR   = 6;      // decomposition stride
constexpr int REC     = 25;     // receptive pad
constexpr int L0      = 16434;  // length after pad + first conv3 (L + 2*REC + r - 2, r=2)

constexpr int TILE_P  = 128;    // positions per workgroup in WMMA kernels (8 waves x 16)
constexpr int CT      = 84;     // Legendre coefficient columns per chunk

// packed A-fragment sizes (halves)
constexpr int FRAG_H   = 32 * 16;                  // one 16x32 f16 fragment, lane-major
constexpr int PACKA_H  = NB * 2 * 3 * 2 * FRAG_H;  // [blk][a/b][k][rt][lane][half]
constexpr int PACKH_H  = 8 * FRAG_H;               // [rt][lane][half]

typedef __attribute__((ext_vector_type(16))) _Float16 v16h;
typedef __attribute__((ext_vector_type(8)))  float    v8f;

// Branchless sigmoid-form GELU entirely on the transcendental pipe:
//   0.5*(1+tanh(u)) == sigma(2u)  =>  gelu(x) = x * rcp(1 + e^{-2u})
//   u = sqrt(2/pi) * (x + 0.044715 x^3)
// v_exp_f32 + v_rcp_f32 (TRANS, co-executes with VALU); saturates via
// exp over/underflow, so no branches and no IEEE div expansion.
__device__ __forceinline__ float gelu_f(float x) {
    float u = 0.79788456080286535588f * x * (1.0f + 0.044715f * x * x);
    float e = __expf(-2.0f * u);
    return x * __builtin_amdgcn_rcpf(1.0f + e);
}

// ---------------------------------------------------------------------------
// Kernel 0: pack all WMMA A-fragments (weights) into their exact lane/half
// VGPR layout so hot kernels fetch a whole v16h fragment with one 32B load.
// A-layout used here (contiguous k, matched by B): half slot h -> K = h + 16*(lane>>4).
// ---------------------------------------------------------------------------
__global__ __launch_bounds__(256)
void pack_weights_kernel(const float* __restrict__ conv_a,  // [NB][CH][CH][3]
                         const float* __restrict__ conv_b,  // [NB][CH][CH][3]
                         const float* __restrict__ w11,     // [128][CH][1]
                         _Float16* __restrict__ packA,      // [NB][2][3][2][32][16]
                         _Float16* __restrict__ packH)      // [8][32][16]
{
    int idx = blockIdx.x * 256 + threadIdx.x;
    if (idx < PACKA_H) {
        int h    = idx & 15;
        int lane = (idx >> 4) & 31;
        int frag = idx >> 9;           // ((blk*2 + ab)*3 + k)*2 + rt
        int rt   = frag & 1;
        int t    = frag >> 1;
        int k    = t % 3;  t /= 3;
        int ab   = t & 1;
        int blk  = t >> 1;
        int m = rt * 16 + (lane & 15);
        int K = h + 16 * (lane >> 4);
        const float* w = (ab ? conv_b : conv_a) + (size_t)blk * CH * CH * 3;
        float v = 0.f;
        if (m < CH && K < CH) v = w[(m * CH + K) * 3 + k];
        packA[idx] = (_Float16)v;
    } else if (idx < PACKA_H + PACKH_H) {
        int i2   = idx - PACKA_H;
        int h    = i2 & 15;
        int lane = (i2 >> 4) & 31;
        int rt   = i2 >> 9;            // 0..7
        int m = rt * 16 + (lane & 15);
        int K = h + 16 * (lane >> 4);
        packH[i2] = (_Float16)((K < CH) ? w11[m * CH + K] : 0.f);
    }
}

// ---------------------------------------------------------------------------
// Kernel 1: circular pad + first conv (10 -> 20 channels, kernel 3)
// x0[b,c,j] = sum_{i,k} w_first[c,i,k] * input[b,i,(j+k-REC) mod L]
// ---------------------------------------------------------------------------
__global__ __launch_bounds__(256)
void first_conv_kernel(const float* __restrict__ xin,   // [B][IN_LEN][L]
                       const float* __restrict__ wf,    // [CH][IN_LEN][3]
                       float* __restrict__ y)           // [B][CH][L0]
{
    long idx = (long)blockIdx.x * 256 + threadIdx.x;
    const long total = (long)B_N * CH * L0;
    if (idx >= total) return;
    int j = (int)(idx % L0);
    long r = idx / L0;
    int c = (int)(r % CH);
    int b = (int)(r / CH);

    const float* xb = xin + (size_t)b * IN_LEN * L_IN;
    float s = 0.f;
#pragma unroll
    for (int k = 0; k < 3; ++k) {
        int src = j + k - REC;
        if (src < 0) src += L_IN;
        if (src >= L_IN) src -= L_IN;
#pragma unroll
        for (int i = 0; i < IN_LEN; ++i)
            s += wf[(c * IN_LEN + i) * 3 + k] * xb[(size_t)i * L_IN + src];
    }
    y[((size_t)b * CH + c) * L0 + j] = s;
}

// ---------------------------------------------------------------------------
// Kernel 2: 20x20 channel conv3 as WMMA GEMM (3 shifted K=32 GEMMs)
// y[b,o,p] = sum_{i,k} w[o,i,k] * x[b,i,p+k] ; optional fused GELU
// wpack points at this conv's [3][2][32][16] packed A fragments.
// ---------------------------------------------------------------------------
__global__ __launch_bounds__(256)
void conv3_wmma_kernel(const float* __restrict__ x, int l_in,
                       const _Float16* __restrict__ wpack,
                       float* __restrict__ y, int gelu_out)
{
    const int l_out = l_in - 2;
    const int b    = blockIdx.y;
    const int p0   = blockIdx.x * TILE_P;
    const int tid  = threadIdx.x;
    const int lane = tid & 31;
    const int wave = tid >> 5;

    // position-major f16 staging: xs[pos][ch], 130 x 32
    __shared__ __align__(32) _Float16 xs[(TILE_P + 2) * CHP];

    const float* xb = x + (size_t)b * CH * l_in;
    // position-fastest float2 staging: coalesced 8B lane loads (rows are 8B aligned)
    constexpr int PCH = (TILE_P + 2) / 2;   // 65 two-position chunks
    for (int task = tid; task < CHP * PCH; task += 256) {
        int c  = task / PCH;
        int pc = task % PCH;
        int pos = pc * 2;
        float v0 = 0.f, v1 = 0.f;
        if (c < CH) {
            int gp = p0 + pos;
            const float* src = &xb[(size_t)c * l_in + gp];
            if (gp + 1 < l_in) {
                float2 v = *(const float2*)src;
                v0 = v.x; v1 = v.y;
            } else if (gp < l_in) {
                v0 = src[0];
            }
        }
        xs[pos * CHP + c]       = (_Float16)v0;
        xs[(pos + 1) * CHP + c] = (_Float16)v1;
    }
    // hint the next tile into cache (global_prefetch_b8)
    if (tid == 0) {
        int np = p0 + TILE_P; if (np >= l_in) np = l_in - 1;
        __builtin_prefetch(&xb[np], 0, 1);
    }
    __syncthreads();

    // A fragments: one aligned 32B load per fragment from the packed layout
    v16h afrag[3][2];
#pragma unroll
    for (int k = 0; k < 3; ++k)
#pragma unroll
        for (int rt = 0; rt < 2; ++rt)
            afrag[k][rt] = *(const v16h*)(wpack + ((k * 2 + rt) * 32 + lane) * 16);

    // B fragments from LDS + WMMA accumulate over the 3 tap shifts
    const int n  = lane & 15;
    const int kg = lane >> 4;
    const int colbase = wave * 16;
    v8f acc0 = {}; v8f acc1 = {};
#pragma unroll
    for (int k = 0; k < 3; ++k) {
        const _Float16* src = &xs[(colbase + n + k) * CHP + kg * 16];
        v16h bfrag = *(const v16h*)src;   // 2x ds_load_b128
        acc0 = __builtin_amdgcn_wmma_f32_16x16x32_f16(
            false, afrag[k][0], false, bfrag, (short)0, acc0, false, false);
        acc1 = __builtin_amdgcn_wmma_f32_16x16x32_f16(
            false, afrag[k][1], false, bfrag, (short)0, acc1, false, false);
    }

    // D layout: lane holds column n; VGPR v -> row M = v + 8*kg (+16 for tile 1)
    int gp = p0 + colbase + n;
    if (gp < l_out) {
        float* yb = y + (size_t)b * CH * l_out;
#pragma unroll
        for (int v = 0; v < 8; ++v) {
            int m0 = v + 8 * kg;
            if (m0 < CH) {
                float r0 = acc0[v];
                if (gelu_out) r0 = gelu_f(r0);
                yb[(size_t)m0 * l_out + gp] = r0;
            }
            int m1 = 16 + m0;
            if (m1 < CH) {
                float r1 = acc1[v];
                if (gelu_out) r1 = gelu_f(r1);
                yb[(size_t)m1 * l_out + gp] = r1;
            }
        }
    }
}

// ---------------------------------------------------------------------------
// Kernel 3: Legendre decomposition + 6x6 mode mix + overlap-add reconstruct,
// fused with residual add + GELU:
//   xn[b,c,j] = gelu(rec[j+6] + lin[b,c,j+4]),  j in [0, l-12)
// ---------------------------------------------------------------------------
__global__ __launch_bounds__(256)
void legendre_combine_kernel(const float* __restrict__ x, int l,
                             const float* __restrict__ lin,   // [B][CH][l-4]
                             const float* __restrict__ linm,  // [CH][M][M] (this block)
                             const float* __restrict__ fd,    // [M][1][12]
                             const float* __restrict__ fr,    // [M][1][12]
                             float* __restrict__ xn)          // [B][CH][l-12]
{
    const int bc  = blockIdx.y;
    const int b   = bc / CH;
    const int c   = bc % CH;
    const int ll  = (l - N_POLY) / S_STR + 1;
    const int t0  = blockIdx.x * CT;
    const int tid = threadIdx.x;

    constexpr int XW = 6 * CT + 18;       // staged x samples
    constexpr int CW = CT + 2;            // staged coefficient columns (t0-1 .. t0+CT)

    __shared__ float xsh[XW];
    __shared__ float lx [CW * M_MODES];
    __shared__ float lx2[CW * M_MODES];
    __shared__ float fdk[M_MODES * N_POLY];
    __shared__ float frk[M_MODES * N_POLY];
    __shared__ float wm [M_MODES * M_MODES];

    const float* xrow = x + ((size_t)b * CH + c) * l;
    const int xbase = 6 * (t0 - 1);
    for (int i = tid; i < XW; i += 256) {
        int gi = xbase + i;
        xsh[i] = (gi >= 0 && gi < l) ? xrow[gi] : 0.f;
    }
    if (tid < M_MODES * N_POLY) { fdk[tid] = fd[tid] * 0.5f; frk[tid] = fr[tid] * 0.5f; }
    if (tid < M_MODES * M_MODES) wm[tid] = linm[c * M_MODES * M_MODES + tid];
    __syncthreads();

    // decomposition coefficients
    for (int idx = tid; idx < CW * M_MODES; idx += 256) {
        int tt = idx / M_MODES, m = idx % M_MODES;
        int t = t0 - 1 + tt;
        float s = 0.f;
        if (t >= 0 && t < ll) {
            int off = 6 * tt;
#pragma unroll
            for (int q = 0; q < N_POLY; ++q) s += fdk[m * N_POLY + q] * xsh[off + q];
        }
        lx[idx] = s;
    }
    __syncthreads();

    // 6x6 mode mix (x2 scale from reference)
    for (int idx = tid; idx < CW * M_MODES; idx += 256) {
        int tt = idx / M_MODES, o = idx % M_MODES;
        float s = 0.f;
#pragma unroll
        for (int i = 0; i < M_MODES; ++i) s += wm[o * M_MODES + i] * lx[tt * M_MODES + i];
        lx2[idx] = 2.f * s;
    }
    __syncthreads();

    // overlap-add reconstruction + residual + gelu
    const float* linrow = lin + ((size_t)b * CH + c) * (l - 4);
    float* xnrow = xn + ((size_t)b * CH + c) * (l - 12);
    for (int i = tid; i < 6 * CT; i += 256) {
        int p = 6 * t0 + i;
        if (p < 6 || p >= l - 6) continue;
        int tA = p / 6;
        float rec = 0.f;
        if (tA < ll) {
            int tt  = tA - t0 + 1;
            int off = p - 6 * tA;                 // [0,6)
#pragma unroll
            for (int m = 0; m < M_MODES; ++m)
                rec += lx2[tt * M_MODES + m] * frk[m * N_POLY + off];
        }
        int tBm = tA - 1;
        if (tBm >= 0 && tBm < ll) {
            int tt  = tBm - t0 + 1;
            int off = p - 6 * tBm;                // [6,12)
#pragma unroll
            for (int m = 0; m < M_MODES; ++m)
                rec += lx2[tt * M_MODES + m] * frk[m * N_POLY + off];
        }
        xnrow[p - 6] = gelu_f(rec + linrow[p - 2]);
    }
}

// ---------------------------------------------------------------------------
// Kernel 4: head — out[b,j] = sum_h wout[h] * gelu(sum_c w11[h,c] * x[b,c,j])
// 128x20 GEMM via 8 WMMA row tiles (K padded 20->32), GELU on D fragments,
// cross-lane reduction with __shfl_xor over the two 16-lane halves.
// ---------------------------------------------------------------------------
__global__ __launch_bounds__(256)
void head_wmma_kernel(const float* __restrict__ x, int l_x,   // [B][CH][l_x]
                      const _Float16* __restrict__ hpack,     // [8][32][16]
                      const float* __restrict__ wout,         // [1][128][1]
                      float* __restrict__ out, int l_out)     // [B][l_out]
{
    const int b    = blockIdx.y;
    const int p0   = blockIdx.x * TILE_P;
    const int tid  = threadIdx.x;
    const int lane = tid & 31;
    const int wave = tid >> 5;

    __shared__ __align__(32) _Float16 xs[TILE_P * CHP];
    __shared__ float wo[128];

    const float* xb = x + (size_t)b * CH * l_x;
    constexpr int PCH = TILE_P / 2;   // 64 two-position chunks
    for (int task = tid; task < CHP * PCH; task += 256) {
        int c  = task / PCH;
        int pc = task % PCH;
        int pos = pc * 2;
        float v0 = 0.f, v1 = 0.f;
        if (c < CH) {
            int gp = p0 + pos;
            const float* src = &xb[(size_t)c * l_x + gp];
            if (gp + 1 < l_x) {
                float2 v = *(const float2*)src;
                v0 = v.x; v1 = v.y;
            } else if (gp < l_x) {
                v0 = src[0];
            }
        }
        xs[pos * CHP + c]       = (_Float16)v0;
        xs[(pos + 1) * CHP + c] = (_Float16)v1;
    }
    if (tid < 128) wo[tid] = wout[tid];
    __syncthreads();

    const int n  = lane & 15;
    const int kg = lane >> 4;
    const int colbase = wave * 16;
    v16h bfrag = *(const v16h*)&xs[(colbase + n) * CHP + kg * 16];

    float partial = 0.f;
#pragma unroll
    for (int rt = 0; rt < 8; ++rt) {
        v16h a = *(const v16h*)(hpack + (rt * 32 + lane) * 16);
        v8f acc = {};
        acc = __builtin_amdgcn_wmma_f32_16x16x32_f16(
            false, a, false, bfrag, (short)0, acc, false, false);
#pragma unroll
        for (int v = 0; v < 8; ++v) {
            int hrow = rt * 16 + v + 8 * kg;
            partial += wo[hrow] * gelu_f(acc[v]);
        }
    }
    partial += __shfl_xor(partial, 16, 32);   // combine the two M halves of column n
    int gp = p0 + colbase + n;
    if (lane < 16 && gp < l_out)
        out[(size_t)b * l_out + gp] = partial;
}

// ---------------------------------------------------------------------------
// Launch orchestration. Workspace: 3 activation buffers of B*CH*L0 floats
// (~42 MB each) + ~56 KB of packed f16 weight fragments.
// ---------------------------------------------------------------------------
extern "C" void kernel_launch(void* const* d_in, const int* in_sizes, int n_in,
                              void* d_out, int out_size, void* d_ws, size_t ws_size,
                              hipStream_t stream) {
    (void)in_sizes; (void)n_in; (void)out_size; (void)ws_size;

    const float* input   = (const float*)d_in[0];
    const float* w_first = (const float*)d_in[1];
    const float* conv_a  = (const float*)d_in[2];  // [NB][CH][CH][3]
    const float* conv_b  = (const float*)d_in[3];  // [NB][CH][CH][3]
    const float* lin_m   = (const float*)d_in[4];  // [NB][CH][M][M]
    const float* w11     = (const float*)d_in[5];  // [128][CH][1]
    const float* w_out   = (const float*)d_in[6];  // [1][128][1]
    const float* filt_d  = (const float*)d_in[7];  // [M][1][12]
    const float* filt_r  = (const float*)d_in[8];  // [M][1][12]
    float* out = (float*)d_out;

    const size_t actElems = (size_t)B_N * CH * L0;
    float* pbuf[2];
    pbuf[0] = (float*)d_ws;
    pbuf[1] = pbuf[0] + actElems;
    float* plin = pbuf[1] + actElems;
    _Float16* packA = (_Float16*)(plin + actElems);   // 32B-aligned (offset % 32 == 0)
    _Float16* packH = packA + PACKA_H;

    // 0) pack WMMA A-fragments (weights) once
    {
        int total = PACKA_H + PACKH_H;
        pack_weights_kernel<<<(total + 255) / 256, 256, 0, stream>>>(
            conv_a, conv_b, w11, packA, packH);
    }

    // 1) pad + first conv -> pbuf[0]
    {
        long total = (long)B_N * CH * L0;
        int grid = (int)((total + 255) / 256);
        first_conv_kernel<<<grid, 256, 0, stream>>>(input, w_first, pbuf[0]);
    }

    // 2) four Legendre blocks; length shrinks by 12 per block
    int l = L0;
    for (int i = 0; i < NB; ++i) {
        float* X = pbuf[i & 1];
        float* T = pbuf[(i + 1) & 1];
        const _Float16* wpA = packA + ((size_t)(i * 2 + 0) * 3 * 2 * FRAG_H);
        const _Float16* wpB = packA + ((size_t)(i * 2 + 1) * 3 * 2 * FRAG_H);
        {   // t = gelu(conv3(x, conv_a[i]))
            int lo = l - 2;
            dim3 g((lo + TILE_P - 1) / TILE_P, B_N);
            conv3_wmma_kernel<<<g, 256, 0, stream>>>(X, l, wpA, T, 1);
        }
        {   // lin = conv3(t, conv_b[i])
            int lo = l - 4;
            dim3 g((lo + TILE_P - 1) / TILE_P, B_N);
            conv3_wmma_kernel<<<g, 256, 0, stream>>>(T, l - 2, wpB, plin, 0);
        }
        {   // Legendre branch + combine -> overwrites T buffer (t is dead)
            int chunks = (l + 6 * CT - 1) / (6 * CT);
            dim3 g(chunks, B_N * CH);
            legendre_combine_kernel<<<g, 256, 0, stream>>>(
                X, l, plin, lin_m + (size_t)i * CH * M_MODES * M_MODES,
                filt_d, filt_r, T);
        }
        l -= 12;   // new activation lives in pbuf[(i+1)&1]
    }

    // 3) head: x4 in pbuf[0] (length 16386), emit first 16384 samples
    {
        dim3 g((L_IN + TILE_P - 1) / TILE_P, B_N);
        head_wmma_kernel<<<g, 256, 0, stream>>>(pbuf[0], l, packH, w_out, out, L_IN);
    }
}
